// WeightedNonLocalBlock_31636729102405
// MI455X (gfx1250) — matrix operand order repaired and checked by reading, not compiled, not run on hardware
//
#include <hip/hip_runtime.h>

// ---------------------------------------------------------------------------
// WeightedNonLocalBlock for MI455X (gfx1250, wave32, WMMA bf16 16x16x32).
//   B=8, C=256, H=W=64, N=4096.
// Pipeline:
//   1) proj_kernel : theta/phi/g = conv1x1(x)  -> bf16, theta/phi stored
//      transposed (B,N,C) for attention A/B fragment-friendly loads,
//      g stored (B,C,N) for PV B-fragments.
//   2) attn_kernel : flash attention, per-wave 16 queries x 256 channels,
//      online softmax, double-buffered key tiles staged with
//      GLOBAL_LOAD_ASYNC_TO_LDS (ASYNCcnt) to overlap fill with WMMA.
//   3) out_kernel  : z = (1-w)*x + w*conv1x1(view(y)), fp32 out.
// Workspace: 4 segments of B*N*C bf16 = 64 MB total.
// ---------------------------------------------------------------------------

#define BB 8
#define CC 256
#define NN 4096

typedef __attribute__((ext_vector_type(16))) __bf16 v16bf;
typedef __attribute__((ext_vector_type(8)))  __bf16 v8bf;
typedef __attribute__((ext_vector_type(8)))  float  v8f;
typedef __attribute__((ext_vector_type(4)))  int    v4i;

__device__ __forceinline__ __bf16 f2bf(float f) {
    unsigned u = __builtin_bit_cast(unsigned, f);
    u += 0x7FFFu + ((u >> 16) & 1u);             // round-to-nearest-even
    unsigned short h = (unsigned short)(u >> 16);
    return __builtin_bit_cast(__bf16, h);
}

__device__ __forceinline__ v8f wmma_bf16(v16bf a, v16bf b, v8f c) {
    // v_wmma_f32_16x16x32_bf16: D = A(16x32) * B(32x16) + C(16x16 f32)
    return __builtin_amdgcn_wmma_f32_16x16x32_bf16(
        false, a, false, b, (short)0, c, false, false);
}

// ---- async global->LDS copy (16 bytes per lane), ASYNCcnt-tracked ---------
#if defined(__HIP_DEVICE_COMPILE__) && defined(__gfx1250__) && \
    __has_builtin(__builtin_amdgcn_global_load_async_to_lds_b128)
#define HAVE_ASYNC_LDS 1
#else
#define HAVE_ASYNC_LDS 0
#endif

#if HAVE_ASYNC_LDS
typedef __attribute__((address_space(1))) v4i* as1_v4i_ptr;
typedef __attribute__((address_space(3))) v4i* as3_v4i_ptr;
#endif

__device__ __forceinline__ void async_copy16(const __bf16* g, __bf16* l) {
#if HAVE_ASYNC_LDS
    __builtin_amdgcn_global_load_async_to_lds_b128(
        (as1_v4i_ptr)(g), (as3_v4i_ptr)(l), 0, 0);
#else
    *(uint4*)l = *(const uint4*)g;           // fallback: VGPR round-trip
#endif
}

__device__ __forceinline__ void async_wait0() {
#if defined(__HIP_DEVICE_COMPILE__) && defined(__gfx1250__)
#if __has_builtin(__builtin_amdgcn_s_wait_asynccnt)
    __builtin_amdgcn_s_wait_asynccnt(0);
#else
    asm volatile("s_wait_asynccnt 0x0" ::: "memory");
#endif
#endif
}

// A fragment (16x32, 16-bit): lane holds row M=lane&15.
// K per lane: j=0..7 -> (lane>>4)*8 + j ; j=8..15 -> 16 + (lane>>4)*8 + (j-8).
// p = &tile[row][k_chunk_base]; row-major tile with contiguous K.
__device__ __forceinline__ v16bf ldA(const __bf16* p, int lh) {
    v8bf lo = *(const v8bf*)(p + lh * 8);
    v8bf hi = *(const v8bf*)(p + 16 + lh * 8);
    v16bf r;
#pragma unroll
    for (int i = 0; i < 8; i++) { r[i] = lo[i]; r[i + 8] = hi[i]; }
    return r;
}

// B fragment (32x16, 16-bit): lane holds column N=lane&15,
// K per lane: j=0..7 -> (lane>>4)*8 + j ; j=8..15 -> 16 + (lane>>4)*8 + (j-8).
// p = &tile[n_row][k_chunk_base]; tile laid out [N][K] with contiguous K.
__device__ __forceinline__ v16bf ldB(const __bf16* p, int lh) {
    v8bf lo = *(const v8bf*)(p + lh * 8);
    v8bf hi = *(const v8bf*)(p + 16 + lh * 8);
    v16bf r;
#pragma unroll
    for (int i = 0; i < 8; i++) { r[i] = lo[i]; r[i + 8] = hi[i]; }
    return r;
}

// ---------------------------------------------------------------------------
// Kernel 1: fused theta/phi/g projections.
// Grid (N/32, B), block 256 (8 waves). Each wave: M-range of 32 (2 tiles),
// N tile = 32 (2 subtiles), 3 outputs -> 12 accumulators.
// ---------------------------------------------------------------------------
__global__ __launch_bounds__(256, 1)
void proj_kernel(const float* __restrict__ x,
                 const float* __restrict__ theta_w, const float* __restrict__ theta_b,
                 const float* __restrict__ phi_w,   const float* __restrict__ phi_b,
                 const float* __restrict__ g_w,     const float* __restrict__ g_b,
                 __bf16* __restrict__ thetaT,       // (B,N,C)
                 __bf16* __restrict__ phiT,         // (B,N,C)
                 __bf16* __restrict__ gout)         // (B,C,N)
{
    __shared__ __bf16 sW[3][256][32];   // [proj][m][k-chunk]  48 KB
    __shared__ __bf16 sX[32][32];       // [n][k-chunk]         2 KB

    const int tid  = threadIdx.x;
    const int wave = tid >> 5, lane = tid & 31;
    const int lh   = lane >> 4, l15 = lane & 15;
    const int b     = blockIdx.y;
    const int nbase = blockIdx.x * 32;
    const int mb    = wave * 32;

    const float* Wsrc[3] = {theta_w, phi_w, g_w};

    v8f acc[3][2][2];
#pragma unroll
    for (int p = 0; p < 3; p++)
#pragma unroll
        for (int mt = 0; mt < 2; mt++)
#pragma unroll
            for (int nt = 0; nt < 2; nt++)
#pragma unroll
                for (int i = 0; i < 8; i++) acc[p][mt][nt][i] = 0.f;

    for (int kc = 0; kc < 8; kc++) {        // K = 256 in chunks of 32
        __syncthreads();
        // stage weights: thread tid loads row m=tid, 32 k values, per proj
#pragma unroll
        for (int p = 0; p < 3; p++) {
            const float* wp = Wsrc[p] + (size_t)tid * CC + kc * 32;
#pragma unroll
            for (int i = 0; i < 8; i++) {
                float4 v = *(const float4*)(wp + i * 4);
                sW[p][tid][i * 4 + 0] = f2bf(v.x);
                sW[p][tid][i * 4 + 1] = f2bf(v.y);
                sW[p][tid][i * 4 + 2] = f2bf(v.z);
                sW[p][tid][i * 4 + 3] = f2bf(v.w);
            }
        }
        // stage x tile transposed: [n][k] (k = input channel)
        {
            int c = tid >> 3, n0 = (tid & 7) * 4;
            const float* xp = x + ((size_t)b * CC + kc * 32 + c) * NN + nbase + n0;
            float4 v = *(const float4*)xp;
            sX[n0 + 0][c] = f2bf(v.x);
            sX[n0 + 1][c] = f2bf(v.y);
            sX[n0 + 2][c] = f2bf(v.z);
            sX[n0 + 3][c] = f2bf(v.w);
        }
        __syncthreads();

        v16bf b0 = ldB(&sX[l15][0], lh);
        v16bf b1 = ldB(&sX[16 + l15][0], lh);
#pragma unroll
        for (int p = 0; p < 3; p++)
#pragma unroll
            for (int mt = 0; mt < 2; mt++) {
                v16bf a = ldA(&sW[p][mb + mt * 16 + l15][0], lh);
                acc[p][mt][0] = wmma_bf16(a, b0, acc[p][mt][0]);
                acc[p][mt][1] = wmma_bf16(a, b1, acc[p][mt][1]);
            }
    }

    const float* Bsrc[3] = {theta_b, phi_b, g_b};
#pragma unroll
    for (int p = 0; p < 3; p++)
#pragma unroll
        for (int mt = 0; mt < 2; mt++)
#pragma unroll
            for (int nt = 0; nt < 2; nt++)
#pragma unroll
                for (int r = 0; r < 8; r++) {
                    int m = mb + mt * 16 + r + 8 * lh;   // C/D layout: M=r(+8)
                    int n = nbase + nt * 16 + l15;
                    __bf16 h = f2bf(acc[p][mt][nt][r] + Bsrc[p][m]);
                    if (p == 0)      thetaT[((size_t)b * NN + n) * CC + m] = h;
                    else if (p == 1) phiT  [((size_t)b * NN + n) * CC + m] = h;
                    else             gout  [((size_t)b * CC + m) * NN + n] = h;
                }
}

// ---------------------------------------------------------------------------
// Kernel 2: flash attention. Grid (N/128, B), block 256 (8 waves).
// Wave owns 16 queries x 256 channels. Key loop: 128 tiles of 32 keys,
// double-buffered in LDS via async global->LDS copies.
// ---------------------------------------------------------------------------
__global__ __launch_bounds__(256, 1)
void attn_kernel(const __bf16* __restrict__ thetaT,  // (B,N,C)
                 const __bf16* __restrict__ phiT,    // (B,N,C)
                 const __bf16* __restrict__ gin,     // (B,C,N)
                 __bf16* __restrict__ yout)          // (B,N,C)
{
    extern __shared__ __bf16 smem[];
    __bf16* sTheta = smem;                       // [8][16][256]       64 KB
    __bf16* sPhi0  = sTheta + 8 * 16 * 256;      // [32 key][256] x2   32 KB
    __bf16* sPhi1  = sPhi0 + 32 * 256;
    __bf16* sV0    = sPhi1 + 32 * 256;           // [256 c][32]   x2   32 KB
    __bf16* sV1    = sV0 + 256 * 32;
    __bf16* sP     = sV1 + 256 * 32;             // [8][16][32]         8 KB

    const int tid  = threadIdx.x;
    const int wave = tid >> 5, lane = tid & 31;
    const int lh   = lane >> 4, l15 = lane & 15;
    const int b    = blockIdx.y;
    const int q0   = blockIdx.x * 128 + wave * 16;

    __bf16* sThw = sTheta + wave * (16 * 256);
    __bf16* sPw  = sP + wave * (16 * 32);

    // per-thread staging addresses for the phi / V key tiles
    const int prow = tid >> 3;                 // phi: key row 0..31
    const int pch  = (tid & 7) * 32;           // phi: 4x16B chunks in row
    const __bf16* phiRow = phiT + ((size_t)b * NN + prow) * CC + pch;
    const __bf16* vRow   = gin + ((size_t)b * CC + tid) * NN;   // V: c row = tid

    // stage this wave's 16 query rows (contiguous 16x256 bf16 region)
    {
        const __bf16* src = thetaT + ((size_t)b * NN + q0) * CC;
        for (int i = lane; i < 512; i += 32)
            async_copy16(src + i * 8, sThw + i * 8);
    }
    // stage key tile 0 into buffer 0
    {
        const __bf16* sp = phiRow;              // keybase 0
        __bf16* dp = sPhi0 + prow * 256 + pch;
#pragma unroll
        for (int i = 0; i < 4; i++) async_copy16(sp + i * 8, dp + i * 8);
        const __bf16* sv = vRow;
        __bf16* dv = sV0 + tid * 32;
#pragma unroll
        for (int i = 0; i < 4; i++) async_copy16(sv + i * 8, dv + i * 8);
    }
    async_wait0();
    __syncthreads();

    float mstate[8], lstate[8];
#pragma unroll
    for (int r = 0; r < 8; r++) { mstate[r] = -1e30f; lstate[r] = 0.f; }
    v8f acc[16];
#pragma unroll
    for (int ct = 0; ct < 16; ct++)
#pragma unroll
        for (int i = 0; i < 8; i++) acc[ct][i] = 0.f;

    for (int kb = 0; kb < 128; kb++) {
        const int cur = kb & 1;
        __bf16* sPhiC = cur ? sPhi1 : sPhi0;
        __bf16* sVC   = cur ? sV1 : sV0;

        // prefetch next key tile into the other buffer (async, no VGPR dest)
        if (kb + 1 < 128) {
            const int nk = (kb + 1) * 32;
            __bf16* sPhiN = cur ? sPhi0 : sPhi1;
            __bf16* sVN   = cur ? sV0 : sV1;
            const __bf16* sp = phiRow + (size_t)nk * CC;
            __bf16* dp = sPhiN + prow * 256 + pch;
#pragma unroll
            for (int i = 0; i < 4; i++) async_copy16(sp + i * 8, dp + i * 8);
            const __bf16* sv = vRow + nk;
            __bf16* dv = sVN + tid * 32;
#pragma unroll
            for (int i = 0; i < 4; i++) async_copy16(sv + i * 8, dv + i * 8);
        }

        // scores S(16q x 32keys) = thetaT_tile * phi_tile, K=C=256
        v8f s0, s1;
#pragma unroll
        for (int i = 0; i < 8; i++) { s0[i] = 0.f; s1[i] = 0.f; }
#pragma unroll
        for (int cc = 0; cc < 8; cc++) {
            const int cb = cc * 32;
            v16bf a   = ldA(sThw + l15 * 256 + cb, lh);
            v16bf bb0 = ldB(sPhiC + l15 * 256 + cb, lh);
            v16bf bb1 = ldB(sPhiC + (16 + l15) * 256 + cb, lh);
            s0 = wmma_bf16(a, bb0, s0);
            s1 = wmma_bf16(a, bb1, s1);
        }

        // online softmax; row = query = r + 8*lh lives in one 16-lane half
        float scale[8];
#pragma unroll
        for (int r = 0; r < 8; r++) {
            float t = fmaxf(s0[r], s1[r]);
            t = fmaxf(t, __shfl_xor(t, 1));
            t = fmaxf(t, __shfl_xor(t, 2));
            t = fmaxf(t, __shfl_xor(t, 4));
            t = fmaxf(t, __shfl_xor(t, 8));
            float mi = fmaxf(mstate[r], t);
            scale[r] = __expf(mstate[r] - mi);
            mstate[r] = mi;
            float p0 = __expf(s0[r] - mi);
            float p1 = __expf(s1[r] - mi);
            float rs = p0 + p1;
            rs += __shfl_xor(rs, 1);
            rs += __shfl_xor(rs, 2);
            rs += __shfl_xor(rs, 4);
            rs += __shfl_xor(rs, 8);
            lstate[r] = lstate[r] * scale[r] + rs;
            sPw[(r + 8 * lh) * 32 + l15]      = f2bf(p0);
            sPw[(r + 8 * lh) * 32 + 16 + l15] = f2bf(p1);
        }
#pragma unroll
        for (int ct = 0; ct < 16; ct++)
#pragma unroll
            for (int r = 0; r < 8; r++) acc[ct][r] *= scale[r];

        // PV: y(16q x 256c) += P(16x32) * V(32 x 256)
        v16bf pa = ldA(sPw + l15 * 32, lh);
#pragma unroll
        for (int ct = 0; ct < 16; ct++) {
            v16bf vb = ldB(sVC + (ct * 16 + l15) * 32, lh);
            acc[ct] = wmma_bf16(pa, vb, acc[ct]);
        }

        // next tile's async fills must land; all waves done with buffers
        async_wait0();
        __syncthreads();
    }

    float rinv[8];
#pragma unroll
    for (int r = 0; r < 8; r++) rinv[r] = 1.0f / lstate[r];
#pragma unroll
    for (int ct = 0; ct < 16; ct++)
#pragma unroll
        for (int r = 0; r < 8; r++) {
            int q = q0 + r + 8 * lh;
            int c = ct * 16 + l15;
            yout[((size_t)b * NN + q) * CC + c] = f2bf(acc[ct][r] * rinv[r]);
        }
}

// ---------------------------------------------------------------------------
// Kernel 3: z = (1-w)*x + w*(wz_w @ view(y) + wz_b).
// view(y): y is (B,N,C); viewed as (B,C,H,W):
//   y_view[b][c][n'] = y[b][ c*16 + (n'>>8) ][ n' & 255 ]
// Grid (N/32, B), block 256.
// ---------------------------------------------------------------------------
__global__ __launch_bounds__(256, 1)
void out_kernel(const float* __restrict__ x,
                const float* __restrict__ wz_w, const float* __restrict__ wz_b,
                const float* __restrict__ wscalar,
                const __bf16* __restrict__ yin,   // (B,N,C)
                float* __restrict__ out)
{
    __shared__ __bf16 sWz[256][32];  // 16 KB
    __shared__ __bf16 sY[32][32];    //  2 KB  [n'][c]

    const int tid  = threadIdx.x;
    const int wave = tid >> 5, lane = tid & 31;
    const int lh   = lane >> 4, l15 = lane & 15;
    const int b     = blockIdx.y;
    const int nbase = blockIdx.x * 32;
    const int mb    = wave * 32;

    v8f acc[2][2];
#pragma unroll
    for (int mt = 0; mt < 2; mt++)
#pragma unroll
        for (int nt = 0; nt < 2; nt++)
#pragma unroll
            for (int i = 0; i < 8; i++) acc[mt][nt][i] = 0.f;

    for (int kc = 0; kc < 8; kc++) {
        __syncthreads();
        {
            const float* wp = wz_w + (size_t)tid * CC + kc * 32;
#pragma unroll
            for (int i = 0; i < 8; i++) {
                float4 v = *(const float4*)(wp + i * 4);
                sWz[tid][i * 4 + 0] = f2bf(v.x);
                sWz[tid][i * 4 + 1] = f2bf(v.y);
                sWz[tid][i * 4 + 2] = f2bf(v.z);
                sWz[tid][i * 4 + 3] = f2bf(v.w);
            }
        }
        {
            int c = tid >> 3, n0 = (tid & 7) * 4;
            int crow = kc * 32 + c;
#pragma unroll
            for (int i = 0; i < 4; i++) {
                int np = nbase + n0 + i;
                int q  = crow * 16 + (np >> 8);
                int ch = np & 255;
                sY[n0 + i][c] = yin[((size_t)b * NN + q) * CC + ch];
            }
        }
        __syncthreads();

        v16bf b0 = ldB(&sY[l15][0], lh);
        v16bf b1 = ldB(&sY[16 + l15][0], lh);
#pragma unroll
        for (int mt = 0; mt < 2; mt++) {
            v16bf a = ldA(&sWz[mb + mt * 16 + l15][0], lh);
            acc[mt][0] = wmma_bf16(a, b0, acc[mt][0]);
            acc[mt][1] = wmma_bf16(a, b1, acc[mt][1]);
        }
    }

    const float wv = wscalar[0];
#pragma unroll
    for (int mt = 0; mt < 2; mt++)
#pragma unroll
        for (int nt = 0; nt < 2; nt++)
#pragma unroll
            for (int r = 0; r < 8; r++) {
                int m  = mb + mt * 16 + r + 8 * lh;
                int np = nbase + nt * 16 + l15;
                size_t idx = ((size_t)b * CC + m) * NN + np;
                out[idx] = (1.0f - wv) * x[idx] + wv * (acc[mt][nt][r] + wz_b[m]);
            }
}

// ---------------------------------------------------------------------------
extern "C" void kernel_launch(void* const* d_in, const int* in_sizes, int n_in,
                              void* d_out, int out_size, void* d_ws, size_t ws_size,
                              hipStream_t stream)
{
    (void)in_sizes; (void)n_in; (void)out_size; (void)ws_size;
    const float* x       = (const float*)d_in[0];
    const float* theta_w = (const float*)d_in[1];
    const float* theta_b = (const float*)d_in[2];
    const float* phi_w   = (const float*)d_in[3];
    const float* phi_b   = (const float*)d_in[4];
    const float* g_w     = (const float*)d_in[5];
    const float* g_b     = (const float*)d_in[6];
    const float* wz_w    = (const float*)d_in[7];
    const float* wz_b    = (const float*)d_in[8];
    const float* wsc     = (const float*)d_in[9];

    __bf16* ws = (__bf16*)d_ws;
    const size_t SEG = (size_t)BB * NN * CC;   // 8M elements (16 MB bf16)
    __bf16* thetaT = ws;
    __bf16* phiT   = ws + SEG;
    __bf16* gws    = ws + 2 * SEG;
    __bf16* yws    = ws + 3 * SEG;

    proj_kernel<<<dim3(NN / 32, BB), 256, 0, stream>>>(
        x, theta_w, theta_b, phi_w, phi_b, g_w, g_b, thetaT, phiT, gws);

    const size_t smem_bytes =
        (size_t)(8 * 16 * 256 + 2 * 32 * 256 + 2 * 256 * 32 + 8 * 16 * 32) *
        sizeof(__bf16);
    attn_kernel<<<dim3(NN / 128, BB), 256, smem_bytes, stream>>>(
        thetaT, phiT, gws, yws);

    out_kernel<<<dim3(NN / 32, BB), 256, 0, stream>>>(
        x, wz_w, wz_b, wsc, yws, (float*)d_out);
}
